// dygcc_dype_v0_Block_54855322305030
// MI455X (gfx1250) — matrix operand order repaired and checked by reading, not compiled, not run on hardware
//
#include <hip/hip_runtime.h>
#include <hip/hip_bf16.h>
#include <hip/hip_fp16.h>

// ---------------------------------------------------------------------------
// Problem constants (from the reference): B=64, C=384, H=W=32, K=32, CR=24
// ---------------------------------------------------------------------------
#define BSZ   64
#define CCH   384
#define C2    192
#define HWPX  1024          // 32*32
#define CRED  24
#define PJ    24576         // 2*C*K
#define MROWS 65536         // B*H*W
#define NHID  1536          // 4*C
#define WELTS (NHID * CCH)  // 589824 weight elements per matrix

typedef __attribute__((ext_vector_type(16))) _Float16 v16h;
typedef __attribute__((ext_vector_type(8)))  float    v8f;

// ---------------------------------------------------------------------------
// CDNA5 async global->LDS staging (ASYNCcnt path), guarded for toolchain.
// Builtin signature (from hipcc diagnostics): param0 is
//   'int __attribute__((vector_size(16))) __device__ *'  (AS1, global src)
// param1 is the LDS destination (AS3). Args 2/3 are imm offset / cpol.
// ---------------------------------------------------------------------------
#if defined(__gfx1250__) && __has_builtin(__builtin_amdgcn_global_load_async_to_lds_b128)
#define USE_ASYNC_LDS 1
#if __has_builtin(__builtin_amdgcn_s_wait_asynccnt)
#define WAIT_ASYNC0() __builtin_amdgcn_s_wait_asynccnt(0)
#else
#define WAIT_ASYNC0() asm volatile("s_wait_asynccnt 0x0" ::: "memory")
#endif
typedef int async_v4i __attribute__((vector_size(16)));
// generic->AS1/AS3 via integer reinterpretation (low 32 bits of a generic LDS
// pointer are the LDS offset per the flat-aperture rules)
#define GLB_AS1(p) ((__attribute__((address_space(1))) async_v4i*)(unsigned long long)(p))
#define LDS_AS3(p) ((__attribute__((address_space(3))) async_v4i*)(unsigned)(unsigned long long)(p))
#else
#define USE_ASYNC_LDS 0
#endif

// ---------------------------------------------------------------------------
// Kernel 0: one-time f32 -> f16 weight conversion (w1: 1536x384, w2: 384x1536)
// ---------------------------------------------------------------------------
__global__ __launch_bounds__(256) void wcvt_kernel(const float* __restrict__ w1,
                                                   const float* __restrict__ w2,
                                                   _Float16* __restrict__ w1h,
                                                   _Float16* __restrict__ w2h) {
  const int base = (blockIdx.x * 256 + threadIdx.x) * 4;
#pragma unroll
  for (int j = 0; j < 4; ++j) {
    w1h[base + j] = (_Float16)w1[base + j];
    w2h[base + j] = (_Float16)w2[base + j];
  }
}

// ---------------------------------------------------------------------------
// Kernel 1: spatial mean pool  p[b,c] = mean_{h,w} x[b,c,h,w]
// ---------------------------------------------------------------------------
__global__ __launch_bounds__(256) void pool_kernel(const float* __restrict__ x,
                                                   float* __restrict__ p) {
  __shared__ float red[256];
  const int bc = blockIdx.x;
  const int t  = threadIdx.x;
  const float* xt = x + (size_t)bc * HWPX;
  float s = xt[t] + xt[t + 256] + xt[t + 512] + xt[t + 768];
  red[t] = s;
  __syncthreads();
  for (int st = 128; st > 0; st >>= 1) {
    if (t < st) red[t] += red[t + st];
    __syncthreads();
  }
  if (t == 0) p[bc] = red[0] * (1.0f / 1024.0f);
}

// ---------------------------------------------------------------------------
// Kernel 2: hyper-network + per-batch feature norm (one block per batch).
// ---------------------------------------------------------------------------
__global__ __launch_bounds__(256) void hyper_kernel(
    const float* __restrict__ p,      const float* __restrict__ kp_w1,
    const float* __restrict__ bn_g,   const float* __restrict__ bn_b,
    const float* __restrict__ bn_m,   const float* __restrict__ bn_v,
    const float* __restrict__ kp_w2,  const float* __restrict__ kp_b2,
    const float* __restrict__ fn_std, const float* __restrict__ fn_mean,
    float* __restrict__ params) {
  __shared__ float sp[CCH];
  __shared__ float sh[CRED];
  __shared__ float red[256];
  __shared__ float red2[256];
  __shared__ float s_u, s_inv;
  const int b = blockIdx.x;
  const int t = threadIdx.x;

  if (t < CCH)        sp[t]       = p[b * CCH + t];
  if (t + 256 < CCH)  sp[t + 256] = p[b * CCH + t + 256];
  __syncthreads();

  if (t < CRED) {
    float s = 0.0f;
    for (int i = 0; i < CCH; ++i) s += kp_w1[t * CCH + i] * sp[i];
    s = (s - bn_m[t]) * rsqrtf(bn_v[t] + 1e-5f) * bn_g[t] + bn_b[t];
    sh[t] = fmaxf(s, 0.0f);
  }
  __syncthreads();

  float sum = 0.0f, sq = 0.0f;
  for (int j = t; j < PJ; j += 256) {
    float r = kp_b2[j];
#pragma unroll
    for (int q = 0; q < CRED; ++q) r += kp_w2[j * CRED + q] * sh[q];
    sum += r; sq += r * r;
  }
  red[t] = sum; red2[t] = sq;
  __syncthreads();
  for (int st = 128; st > 0; st >>= 1) {
    if (t < st) { red[t] += red[t + st]; red2[t] += red2[t + st]; }
    __syncthreads();
  }
  if (t == 0) {
    float u   = red[0] * (1.0f / PJ);
    float var = red2[0] * (1.0f / PJ) - u * u;
    s_u = u; s_inv = rsqrtf(var + 1e-12f);
  }
  __syncthreads();
  const float u = s_u, inv = s_inv;
  float* pb = params + (size_t)b * PJ;
  for (int j = t; j < PJ; j += 256) {
    float r = kp_b2[j];
#pragma unroll
    for (int q = 0; q < CRED; ++q) r += kp_w2[j * CRED + q] * sh[q];
    pb[j] = (r - u) * inv * fn_std[j] + fn_mean[j];
  }
}

// ---------------------------------------------------------------------------
// Kernel 3: positional add + circulant spatial mix + bias. One block per (b,c).
// Writes y in NHWC layout: y_t[(b*1024 + h*32 + w)*384 + c].
// ---------------------------------------------------------------------------
__global__ __launch_bounds__(256) void mix_kernel(const float* __restrict__ x,
                                                  const float* __restrict__ params,
                                                  const float* __restrict__ bias,
                                                  float* __restrict__ y_t) {
  __shared__ float xin[32][33];
  __shared__ float flt[32];
  const int bc = blockIdx.x;
  const int b  = bc / CCH;
  const int c  = bc % CCH;
  const int t  = threadIdx.x;
  const float* pb = params + (size_t)b * PJ;
  const bool hbr = (c < C2);
  const int  c2  = hbr ? c : (c - C2);

  if (t < 32) {
    const int fi = hbr ? ((2 * C2 + c2) * 32 + t)      // Hw chunk
                       : ((3 * C2 + c2) * 32 + t);     // Wwt chunk
    flt[t] = pb[fi];
  }
  const float* xt = x + ((size_t)b * CCH + c) * HWPX;
#pragma unroll
  for (int i = 0; i < 4; ++i) {
    const int idx = t + i * 256;
    const int r = idx >> 5, col = idx & 31;
    const float pe = hbr ? pb[c2 * 32 + r]              // H_pe: varies with row g
                         : pb[(C2 + c2) * 32 + col];    // W_pe: varies with col v
    xin[r][col] = xt[idx] + pe;
  }
  __syncthreads();

  const float bv = bias[c];
#pragma unroll
  for (int i = 0; i < 4; ++i) {
    const int idx = t + i * 256;
    const int h = idx >> 5, w = idx & 31;
    float acc = bv;
    if (hbr) {
#pragma unroll
      for (int g = 0; g < 32; ++g) acc += flt[(g - h) & 31] * xin[g][w];
    } else {
#pragma unroll
      for (int v = 0; v < 32; ++v) acc += flt[(v - w) & 31] * xin[h][v];
    }
    y_t[((size_t)b * HWPX + idx) * CCH + c] = acc;
  }
}

// ---------------------------------------------------------------------------
// Kernel 4: channel LayerNorm per NHWC row, emit f16 activations for WMMA.
// ---------------------------------------------------------------------------
__global__ __launch_bounds__(256) void ln_kernel(const float* __restrict__ y_t,
                                                 const float* __restrict__ ln_w,
                                                 const float* __restrict__ ln_b,
                                                 _Float16* __restrict__ a1) {
  const int wid  = threadIdx.x >> 5;
  const int lane = threadIdx.x & 31;
  const int n    = blockIdx.x * 8 + wid;
  const float* row = y_t + (size_t)n * CCH;
  float v[12];
  float sum = 0.0f, sq = 0.0f;
#pragma unroll
  for (int i = 0; i < 12; ++i) {
    v[i] = row[lane + i * 32];
    sum += v[i]; sq += v[i] * v[i];
  }
#pragma unroll
  for (int m = 16; m > 0; m >>= 1) {
    sum += __shfl_xor(sum, m, 32);
    sq  += __shfl_xor(sq,  m, 32);
  }
  const float mu  = sum * (1.0f / CCH);
  const float inv = rsqrtf(sq * (1.0f / CCH) - mu * mu + 1e-6f);
  _Float16* orow = a1 + (size_t)n * CCH;
#pragma unroll
  for (int i = 0; i < 12; ++i) {
    const int cidx = lane + i * 32;
    orow[cidx] = (_Float16)((v[i] - mu) * inv * ln_w[cidx] + ln_b[cidx]);
  }
}

// ---------------------------------------------------------------------------
// WMMA fragment builders (ISA §7.12.2 layouts, wave32).
// ---------------------------------------------------------------------------
__device__ inline v16h frag_a(const _Float16* row, int half) {
  const uint4* p = (const uint4*)row;     // 4 x 16B per 64B row
  v16h a;
  ((uint4*)&a)[0] = p[half];              // K = half*8 .. half*8+7
  ((uint4*)&a)[1] = p[2 + half];          // K = 16+half*8 .. +7
  return a;
}
__device__ inline v16h frag_b(const _Float16* row, int khalf) {
  const uint4* p = (const uint4*)row;
  v16h b;
  ((uint4*)&b)[0] = p[khalf];             // K = khalf*8 .. khalf*8+7
  ((uint4*)&b)[1] = p[2 + khalf];         // K = 16+khalf*8 .. +7
  return b;
}

// ---------------------------------------------------------------------------
// Tile staging: A tile 128x32 f16 (512 x 16B), B tile 64x32 f16 (256 x 16B).
// Async path uses GLOBAL_LOAD_ASYNC_TO_LDS_B128 (ASYNCcnt); fallback bounces
// through VGPRs. Both tiles are f16 in memory (weights pre-converted).
// ---------------------------------------------------------------------------
__device__ inline void stage_tiles(const _Float16* __restrict__ A,
                                   const _Float16* __restrict__ W,
                                   _Float16* __restrict__ Asb,
                                   _Float16* __restrict__ Bsb,
                                   int m0, int n0, int k0, int t, int K) {
#if USE_ASYNC_LDS
#pragma unroll
  for (int i = 0; i < 2; ++i) {
    const int id = t + i * 256;
    const int r = id >> 2, cp = (id & 3) * 8;
    __builtin_amdgcn_global_load_async_to_lds_b128(
        GLB_AS1(A + (size_t)(m0 + r) * K + k0 + cp),
        LDS_AS3(Asb + r * 32 + cp), 0, 0);
  }
  {
    const int r = t >> 2, cp = (t & 3) * 8;
    __builtin_amdgcn_global_load_async_to_lds_b128(
        GLB_AS1(W + (size_t)(n0 + r) * K + k0 + cp),
        LDS_AS3(Bsb + r * 32 + cp), 0, 0);
  }
#else
#pragma unroll
  for (int i = 0; i < 2; ++i) {
    const int id = t + i * 256;
    const int r = id >> 2, cp = id & 3;
    const uint4* src = (const uint4*)(A + (size_t)(m0 + r) * K + k0);
    ((uint4*)(Asb + r * 32))[cp] = src[cp];
  }
  {
    const int r = t >> 2, cp = t & 3;
    const uint4* src = (const uint4*)(W + (size_t)(n0 + r) * K + k0);
    ((uint4*)(Bsb + r * 32))[cp] = src[cp];
  }
  if (k0 + 32 < K)
    __builtin_prefetch(A + (size_t)(m0 + (t >> 1)) * K + k0 + 32, 0, 1);
#endif
}

// ---------------------------------------------------------------------------
// Kernel 5: GEMM1 (65536x384 @ 384->1536) + bias + exact GELU -> f16 hidden.
// BM=128, BN=64, BK=32, double-buffered LDS, 8 waves x (32x32) = 2x2 WMMA.
// ---------------------------------------------------------------------------
__global__ __launch_bounds__(256) void gemm1_kernel(const _Float16* __restrict__ A,
                                                    const _Float16* __restrict__ W, // (1536,384) f16
                                                    const float* __restrict__ b1,
                                                    _Float16* __restrict__ hid) {
  __shared__ __align__(16) _Float16 As[2][128 * 32];
  __shared__ __align__(16) _Float16 Bs[2][64 * 32];
  const int K  = CCH;                 // 384
  const int m0 = blockIdx.y * 128;
  const int n0 = blockIdx.x * 64;
  const int t    = threadIdx.x;
  const int lane = t & 31;
  const int wv   = t >> 5;
  const int wm   = wv >> 1;           // 0..3
  const int wn   = wv & 1;            // 0..1
  const int half = lane >> 4;

  v8f acc[2][2] = {};

  stage_tiles(A, W, As[0], Bs[0], m0, n0, 0, t, K);
  for (int k0 = 0, it = 0; k0 < K; k0 += 32, ++it) {
    const int cur = it & 1;
#if USE_ASYNC_LDS
    WAIT_ASYNC0();
#endif
    __syncthreads();   // publish tile `cur`; retire all readers of tile `cur^1`
    if (k0 + 32 < K)
      stage_tiles(A, W, As[cur ^ 1], Bs[cur ^ 1], m0, n0, k0 + 32, t, K);

    v16h bfr[2];
#pragma unroll
    for (int ni = 0; ni < 2; ++ni)
      bfr[ni] = frag_b(Bs[cur] + (wn * 32 + ni * 16 + (lane & 15)) * 32, half);
#pragma unroll
    for (int mi = 0; mi < 2; ++mi) {
      const v16h afr = frag_a(As[cur] + (wm * 32 + mi * 16 + (lane & 15)) * 32, half);
#pragma unroll
      for (int ni = 0; ni < 2; ++ni)
        acc[mi][ni] = __builtin_amdgcn_wmma_f32_16x16x32_f16(
            false, afr, false, bfr[ni], (short)0, acc[mi][ni], false, false);
    }
  }

#pragma unroll
  for (int mi = 0; mi < 2; ++mi) {
#pragma unroll
    for (int ni = 0; ni < 2; ++ni) {
      const int mb = m0 + wm * 32 + mi * 16 + (lane >> 4) * 8;
      const int nc = n0 + wn * 32 + ni * 16 + (lane & 15);
      const float bn = b1[nc];
#pragma unroll
      for (int r = 0; r < 8; ++r) {
        float v = acc[mi][ni][r] + bn;
        v = 0.5f * v * (1.0f + erff(v * 0.70710678118654752f));
        hid[(size_t)(mb + r) * NHID + nc] = (_Float16)v;
      }
    }
  }
}

// ---------------------------------------------------------------------------
// Kernel 6: GEMM2 (65536x1536 @ 1536->384) + bias + gamma scale + residual.
// ---------------------------------------------------------------------------
__global__ __launch_bounds__(256) void gemm2_kernel(const _Float16* __restrict__ A, // hidden
                                                    const _Float16* __restrict__ W, // (384,1536) f16
                                                    const float* __restrict__ b2,
                                                    const float* __restrict__ gamma,
                                                    const float* __restrict__ x,
                                                    float* __restrict__ out) {
  __shared__ __align__(16) _Float16 As[2][128 * 32];
  __shared__ __align__(16) _Float16 Bs[2][64 * 32];
  const int K  = NHID;                // 1536
  const int m0 = blockIdx.y * 128;
  const int n0 = blockIdx.x * 64;
  const int t    = threadIdx.x;
  const int lane = t & 31;
  const int wv   = t >> 5;
  const int wm   = wv >> 1;
  const int wn   = wv & 1;
  const int half = lane >> 4;

  v8f acc[2][2] = {};

  stage_tiles(A, W, As[0], Bs[0], m0, n0, 0, t, K);
  for (int k0 = 0, it = 0; k0 < K; k0 += 32, ++it) {
    const int cur = it & 1;
#if USE_ASYNC_LDS
    WAIT_ASYNC0();
#endif
    __syncthreads();
    if (k0 + 32 < K)
      stage_tiles(A, W, As[cur ^ 1], Bs[cur ^ 1], m0, n0, k0 + 32, t, K);

    v16h bfr[2];
#pragma unroll
    for (int ni = 0; ni < 2; ++ni)
      bfr[ni] = frag_b(Bs[cur] + (wn * 32 + ni * 16 + (lane & 15)) * 32, half);
#pragma unroll
    for (int mi = 0; mi < 2; ++mi) {
      const v16h afr = frag_a(As[cur] + (wm * 32 + mi * 16 + (lane & 15)) * 32, half);
#pragma unroll
      for (int ni = 0; ni < 2; ++ni)
        acc[mi][ni] = __builtin_amdgcn_wmma_f32_16x16x32_f16(
            false, afr, false, bfr[ni], (short)0, acc[mi][ni], false, false);
    }
  }

#pragma unroll
  for (int mi = 0; mi < 2; ++mi) {
#pragma unroll
    for (int ni = 0; ni < 2; ++ni) {
      const int mb = m0 + wm * 32 + mi * 16 + (lane >> 4) * 8;
      const int c  = n0 + wn * 32 + ni * 16 + (lane & 15);
      const float bb = b2[c];
      const float gg = gamma[c];
#pragma unroll
      for (int r = 0; r < 8; ++r) {
        const int mr = mb + r;
        const int b  = mr >> 10;
        const int hw = mr & 1023;
        const size_t oi = ((size_t)b * CCH + c) * HWPX + hw;
        out[oi] = x[oi] + gg * (acc[mi][ni][r] + bb);
      }
    }
  }
}

// ---------------------------------------------------------------------------
// Host-side launch
// ---------------------------------------------------------------------------
static inline size_t align256(size_t v) { return (v + 255) & ~(size_t)255; }

extern "C" void kernel_launch(void* const* d_in, const int* in_sizes, int n_in,
                              void* d_out, int out_size, void* d_ws, size_t ws_size,
                              hipStream_t stream) {
  const float* x       = (const float*)d_in[0];
  const float* kp_w1   = (const float*)d_in[1];
  const float* bn_g    = (const float*)d_in[2];
  const float* bn_b    = (const float*)d_in[3];
  const float* bn_m    = (const float*)d_in[4];
  const float* bn_v    = (const float*)d_in[5];
  const float* kp_w2   = (const float*)d_in[6];
  const float* kp_b2   = (const float*)d_in[7];
  const float* fn_std  = (const float*)d_in[8];
  const float* fn_mean = (const float*)d_in[9];
  const float* bias    = (const float*)d_in[10];
  const float* ln_w    = (const float*)d_in[11];
  const float* ln_b    = (const float*)d_in[12];
  const float* w1      = (const float*)d_in[13];
  const float* b1      = (const float*)d_in[14];
  const float* w2      = (const float*)d_in[15];
  const float* b2      = (const float*)d_in[16];
  const float* gamma   = (const float*)d_in[17];
  float* out = (float*)d_out;

  char* ws = (char*)d_ws;
  size_t off = 0;
  float*     p      = (float*)(ws + off);     off += align256((size_t)BSZ * CCH * 4);
  float*     params = (float*)(ws + off);     off += align256((size_t)BSZ * PJ * 4);
  float*     y_t    = (float*)(ws + off);     off += align256((size_t)MROWS * CCH * 4);
  _Float16*  a1     = (_Float16*)(ws + off);  off += align256((size_t)MROWS * CCH * 2);
  _Float16*  hid    = (_Float16*)(ws + off);  off += align256((size_t)MROWS * NHID * 2);
  _Float16*  w1h    = (_Float16*)(ws + off);  off += align256((size_t)WELTS * 2);
  _Float16*  w2h    = (_Float16*)(ws + off);  off += align256((size_t)WELTS * 2);
  (void)ws_size; (void)in_sizes; (void)n_in; (void)out_size;

  wcvt_kernel<<<WELTS / 1024, 256, 0, stream>>>(w1, w2, w1h, w2h);
  pool_kernel<<<BSZ * CCH, 256, 0, stream>>>(x, p);
  hyper_kernel<<<BSZ, 256, 0, stream>>>(p, kp_w1, bn_g, bn_b, bn_m, bn_v,
                                        kp_w2, kp_b2, fn_std, fn_mean, params);
  mix_kernel<<<BSZ * CCH, 256, 0, stream>>>(x, params, bias, y_t);
  ln_kernel<<<MROWS / 8, 256, 0, stream>>>(y_t, ln_w, ln_b, a1);
  gemm1_kernel<<<dim3(NHID / 64, MROWS / 128), 256, 0, stream>>>(a1, w1h, b1, hid);
  gemm2_kernel<<<dim3(CCH / 64, MROWS / 128), 256, 0, stream>>>(hid, w2h, b2, gamma, x, out);
}
